// MMDVAE_26061861552780
// MI455X (gfx1250) — matrix operand order repaired and checked by reading, not compiled, not run on hardware
//
#include <hip/hip_runtime.h>
#include <math.h>

// ---------------------------------------------------------------------------
// MMD-VAE forward on MI455X (gfx1250, wave32).
// GEMMs: v_wmma_f32_16x16x32_f16 with double-buffered LDS tiles filled by
// the CDNA5 async copy engine (global_load_async_to_lds_b128 / ASYNCcnt).
// ---------------------------------------------------------------------------

typedef __attribute__((ext_vector_type(16))) _Float16 v16h;
typedef __attribute__((ext_vector_type(8)))  float    v8f;
typedef __attribute__((ext_vector_type(4)))  float    v4f;

#define NROWS   4096
#define IN_DIM  784
#define KP1     800      // 784 padded up to multiple of 32
#define HD      1024
#define OUT3    1568     // 2*IN_DIM
#define LOG2PI  1.8378770664093453f

#define TM 128           // block tile M
#define TN 64            // block tile N
#define TK 32            // K step (WMMA K)
#define LPAD 40          // LDS row stride in halves (32 data + 8 pad => 80B)

union HF { v16h h; v4f f[2]; };

// ---------------------------------------------------------------------------
// CDNA5 async copy: 16B per lane, global -> LDS, tracked by ASYNCcnt.
// ISA 10.2: LDS flat addresses use addr[31:0] as the LDS byte offset, so the
// low 32 bits of a generic __shared__ pointer are the VDST operand.
// ---------------------------------------------------------------------------
__device__ __forceinline__ void async_cp16(const _Float16* gsrc,
                                           const _Float16* ldst) {
  unsigned loff = (unsigned)(uintptr_t)ldst;
  asm volatile("global_load_async_to_lds_b128 %0, %1, off"
               :: "v"(loff), "v"(gsrc)
               : "memory");
}
__device__ __forceinline__ void wait_async0() {
  asm volatile("s_wait_asynccnt 0x0" ::: "memory");
}

// ---- WMMA fragment loaders (layouts per CDNA5 ISA 7.12.2) ------
// A (16x32 f16): lanes 0-15: row M=lane, K kt+0..7 / kt+16..23;
//                lanes 16-31: row M=lane-16, K kt+8..15 / kt+24..31.
__device__ __forceinline__ v16h lds_a_frag(const _Float16* S, int row0, int lane) {
  const int row = row0 + (lane & 15);
  const int kb  = (lane >> 4) << 3;
  const _Float16* p = S + row * LPAD + kb;
  HF u;
  u.f[0] = *(const v4f*)(p);
  u.f[1] = *(const v4f*)(p + 16);
  return u.h;
}
// B (32x16 f16) from pre-transposed weights: lane's column, K kt+8h..+7 / kt+16+8h..+7.
__device__ __forceinline__ v16h lds_b_frag(const _Float16* S, int col0, int lane) {
  const int col = col0 + (lane & 15);
  const int kb  = (lane >> 4) << 3;
  const _Float16* p = S + col * LPAD + kb;
  HF u;
  u.f[0] = *(const v4f*)(p);
  u.f[1] = *(const v4f*)(p + 16);
  return u.h;
}
// Direct-from-global variants (used by the fused dec3+rec kernel).
__device__ __forceinline__ v16h load_a_frag(const _Float16* __restrict__ A,
                                            int lda, int row0, int kt, int lane) {
  const int row = row0 + (lane & 15);
  const int kb  = kt + ((lane >> 4) << 3);
  const _Float16* p = A + (long)row * lda + kb;
  HF u;
  u.f[0] = *(const v4f*)(p);
  u.f[1] = *(const v4f*)(p + 16);
  return u.h;
}
__device__ __forceinline__ v16h load_b_frag(const _Float16* __restrict__ Bt,
                                            int ldb, int col0, int kt, int lane) {
  const int col = col0 + (lane & 15);
  const int kb  = kt + ((lane >> 4) << 3);
  const _Float16* p = Bt + (long)col * ldb + kb;
  HF u;
  u.f[0] = *(const v4f*)(p);
  u.f[1] = *(const v4f*)(p + 16);
  return u.h;
}

__device__ __forceinline__ v8f wmma_f32_16x16x32(v16h a, v16h b, v8f c) {
  return __builtin_amdgcn_wmma_f32_16x16x32_f16(false, a, false, b,
                                                (short)0, c, false, false);
}

// ---------------------------------------------------------------------------
// Prep kernels.
// ---------------------------------------------------------------------------
__global__ __launch_bounds__(256) void k_cvt_x(const float* __restrict__ x,
                                               _Float16* __restrict__ xh) {
  int idx = blockIdx.x * 256 + threadIdx.x;       // over NROWS*KP1
  int i = idx / KP1, k = idx % KP1;
  xh[idx] = (k < IN_DIM) ? (_Float16)x[i * IN_DIM + k] : (_Float16)0.0f;
}

__global__ __launch_bounds__(256) void k_transpose_w(const float* __restrict__ W,
                                                     _Float16* __restrict__ Bt,
                                                     int Kin, int N, int Kp) {
  long idx = (long)blockIdx.x * 256 + threadIdx.x;  // over N*Kp
  int n = (int)(idx / Kp), k = (int)(idx % Kp);
  Bt[idx] = (k < Kin) ? (_Float16)W[(long)k * N + n] : (_Float16)0.0f;
}

__global__ void k_zero3(float* a) { if (threadIdx.x < 3) a[threadIdx.x] = 0.0f; }

// ---------------------------------------------------------------------------
// Fused GEMM + bias + tanh with async-LDS double buffering.
// Block = 256 thr = 8 waves; tile 128x64; wave = 32x32 (2x2 WMMA tiles).
// Per K-step a block stages A (128x32) + B (64x32) in LDS via the async
// engine while computing on the other buffer.
// ---------------------------------------------------------------------------
__global__ __launch_bounds__(256) void k_gemm_tanh(const _Float16* __restrict__ A,
                                                   const _Float16* __restrict__ Bt,
                                                   const float* __restrict__ bias,
                                                   _Float16* __restrict__ Out,
                                                   int N, int Kp) {
  __shared__ _Float16 As[2][TM * LPAD];
  __shared__ _Float16 Bs[2][TN * LPAD];

  const int tid  = threadIdx.x;
  const int lane = tid & 31;
  const int wid  = tid >> 5;
  const int m0 = (wid & 3) * 32;                  // wave M offset in tile
  const int n0 = (wid >> 2) * 32;                 // wave N offset in tile
  const long gM = (long)blockIdx.y * TM;
  const long gN = (long)blockIdx.x * TN;

  // Copy mapping: A tile = 128 rows x 4 x 16B chunks -> 2 chunks/thread
  //               B tile =  64 rows x 4 x 16B chunks -> 1 chunk/thread
  const int ar = tid >> 1;                        // A row 0..127
  const int ac = (tid & 1) * 16;                  // first half-index of 32B pair
  const int br = tid >> 2;                        // B row 0..63
  const int bc = (tid & 3) * 8;                   // half-index of 16B chunk

  const _Float16* gA = A + (gM + ar) * Kp + ac;
  const _Float16* gB = Bt + (gN + br) * Kp + bc;

#define ISSUE_TILE(buf, kt)                                                    \
  {                                                                            \
    async_cp16(gA + (kt),      &As[buf][ar * LPAD + ac]);                      \
    async_cp16(gA + (kt) + 8,  &As[buf][ar * LPAD + ac + 8]);                  \
    async_cp16(gB + (kt),      &Bs[buf][br * LPAD + bc]);                      \
  }

  v8f c00 = {}, c01 = {}, c10 = {}, c11 = {};
  const int T = Kp / TK;

  ISSUE_TILE(0, 0)
  wait_async0();
  __syncthreads();

  for (int t = 0; t < T; ++t) {
    const int cur = t & 1;
    if (t + 1 < T) ISSUE_TILE(1 - cur, (t + 1) * TK)   // overlaps with compute

    const _Float16* Sa = As[cur];
    const _Float16* Sb = Bs[cur];
    v16h a0 = lds_a_frag(Sa, m0,      lane);
    v16h a1 = lds_a_frag(Sa, m0 + 16, lane);
    v16h b0 = lds_b_frag(Sb, n0,      lane);
    v16h b1 = lds_b_frag(Sb, n0 + 16, lane);
    c00 = wmma_f32_16x16x32(a0, b0, c00);
    c01 = wmma_f32_16x16x32(a0, b1, c01);
    c10 = wmma_f32_16x16x32(a1, b0, c10);
    c11 = wmma_f32_16x16x32(a1, b1, c11);

    if (t + 1 < T) wait_async0();   // next tile landed (own waves' portion)
    __syncthreads();                // publish block-wide + protect buffer reuse
  }
#undef ISSUE_TILE

  // C/D layout: VGPR v, lanes 0-15 -> M=v, lanes 16-31 -> M=v+8; N = lane&15.
  const int half = (lane >> 4) << 3;
#define STORE_TILE(C, MT, NT)                                                  \
  {                                                                            \
    int col = (int)gN + (NT) + (lane & 15);                                    \
    int rb  = (int)gM + (MT) + half;                                           \
    float bb = bias[col];                                                      \
    _Pragma("unroll") for (int v = 0; v < 8; ++v)                              \
      Out[(long)(rb + v) * N + col] = (_Float16)tanhf(C[v] + bb);              \
  }
  STORE_TILE(c00, m0,      n0)
  STORE_TILE(c01, m0,      n0 + 16)
  STORE_TILE(c10, m0 + 16, n0)
  STORE_TILE(c11, m0 + 16, n0 + 16)
#undef STORE_TILE
}

// ---------------------------------------------------------------------------
// Encoder head: enc = H2 @ W3 + b3 (1024 -> 2), then z, kl.
// One wave per row. INITIALIZES out[i] = 0.5*kl[i].
// ---------------------------------------------------------------------------
__global__ __launch_bounds__(256) void k_enc3(const _Float16* __restrict__ H2,
                                              const float* __restrict__ W3,
                                              const float* __restrict__ b3,
                                              const float* __restrict__ eps_z,
                                              float* __restrict__ z,
                                              float* __restrict__ out) {
  const int lane = threadIdx.x & 31;
  const int row  = blockIdx.x * 8 + (threadIdx.x >> 5);
  const _Float16* h = H2 + (long)row * HD;
  float a0 = 0.f, a1 = 0.f;
  for (int k = lane; k < HD; k += 32) {
    float hv = (float)h[k];
    a0 += hv * W3[2 * k + 0];
    a1 += hv * W3[2 * k + 1];
  }
  for (int off = 16; off; off >>= 1) {
    a0 += __shfl_xor(a0, off, 32);
    a1 += __shfl_xor(a1, off, 32);
  }
  if (lane == 0) {
    float m  = a0 + b3[0];
    float ls = a1 + b3[1];
    float sd = expf(ls);
    float kl = -ls + 0.5f * (sd * sd + m * m - 1.0f);
    z[row]   = m + sd * eps_z[row];
    out[row] = 0.5f * kl;                 // (1-ALPHA)*kl, ALPHA=0.5
  }
}

// Decoder layer 1 is a rank-1 outer product: elementwise.
__global__ __launch_bounds__(256) void k_dec1(const float* __restrict__ z,
                                              const float* __restrict__ W1,
                                              const float* __restrict__ b1,
                                              _Float16* __restrict__ d1) {
  int idx = blockIdx.x * 256 + threadIdx.x;  // NROWS*HD
  int i = idx >> 10, j = idx & 1023;
  d1[idx] = (_Float16)tanhf(z[i] * W1[j] + b1[j]);
}

// ---------------------------------------------------------------------------
// Fused dec3 + rec_loss (WMMA, direct-global fragments; W3t is L2-resident).
// ---------------------------------------------------------------------------
__global__ __launch_bounds__(256) void k_dec3_rec(const _Float16* __restrict__ D2,
                                                  const _Float16* __restrict__ W3t,
                                                  const float* __restrict__ b3,
                                                  const float* __restrict__ X,
                                                  float* __restrict__ out) {
  const int lane = threadIdx.x & 31;
  const int wid  = threadIdx.x >> 5;
  const int r0 = blockIdx.y * 128 + wid * 16;
  const int j0 = blockIdx.x * 16;

  v8f cm = {}, cl = {};
  for (int kt = 0; kt < HD; kt += 32) {
    v16h a  = load_a_frag(D2, HD, r0, kt, lane);
    v16h bm = load_b_frag(W3t, HD, j0,          kt, lane);
    v16h bl = load_b_frag(W3t, HD, IN_DIM + j0, kt, lane);
    cm = wmma_f32_16x16x32(a, bm, cm);
    cl = wmma_f32_16x16x32(a, bl, cl);
  }
  const int col = j0 + (lane & 15);
  const int rb  = r0 + ((lane >> 4) << 3);
  const float bm_ = b3[col];
  const float bl_ = b3[IN_DIM + col];

  float recsum[8];
#pragma unroll
  for (int v = 0; v < 8; ++v) {
    float xm = cm[v] + bm_;
    float xl = cl[v] + bl_;
    float xv = X[(long)(rb + v) * IN_DIM + col];
    float t  = (xv - xm) * __expf(-xl);
    recsum[v] = 0.5f * t * t + xl + 0.5f * LOG2PI;   // -log_prob
  }
#pragma unroll
  for (int off = 1; off < 16; off <<= 1) {
#pragma unroll
    for (int v = 0; v < 8; ++v) recsum[v] += __shfl_xor(recsum[v], off, 32);
  }
  if ((lane & 15) == 0) {
#pragma unroll
    for (int v = 0; v < 8; ++v) atomicAdd(&out[rb + v], recsum[v]);
  }
}

// ---------------------------------------------------------------------------
// MMD: three 4096x4096 RBF sums (pd=1), LDS-tiled.
// ---------------------------------------------------------------------------
__global__ __launch_bounds__(256) void k_mmd(const float* __restrict__ z,
                                             const float* __restrict__ p,
                                             float* __restrict__ acc) {
  __shared__ float zs[256], ps[256];
  const int i = blockIdx.x * 256 + threadIdx.x;
  const float zi = z[i], pi = p[i];
  float szz = 0.f, spp = 0.f, szp = 0.f;
  for (int t = 0; t < NROWS; t += 256) {
    __syncthreads();
    zs[threadIdx.x] = z[t + threadIdx.x];
    ps[threadIdx.x] = p[t + threadIdx.x];
    __syncthreads();
    for (int j = 0; j < 256; ++j) {
      float dz = zi - zs[j]; szz += __expf(-dz * dz);
      float dp = pi - ps[j]; spp += __expf(-dp * dp);
      float dc = zi - ps[j]; szp += __expf(-dc * dc);
    }
  }
  for (int off = 16; off; off >>= 1) {
    szz += __shfl_xor(szz, off, 32);
    spp += __shfl_xor(spp, off, 32);
    szp += __shfl_xor(szp, off, 32);
  }
  if ((threadIdx.x & 31) == 0) {
    atomicAdd(&acc[0], szz);
    atomicAdd(&acc[1], spp);
    atomicAdd(&acc[2], szp);
  }
}

__global__ __launch_bounds__(256) void k_final(float* __restrict__ out,
                                               const float* __restrict__ acc) {
  int i = blockIdx.x * 256 + threadIdx.x;
  const float inv = 1.0f / ((float)NROWS * (float)NROWS);
  float mmd = (acc[0] + acc[1] - 2.0f * acc[2]) * inv;
  out[i] += 0.5f * mmd;                   // (ALPHA+LAMBD-1)*mmd = 0.5*mmd
}

// ---------------------------------------------------------------------------
// Workspace layout (bytes), total ~49.2 MB.
// ---------------------------------------------------------------------------
#define WS_XH    0UL
#define WS_W1T   6553600UL
#define WS_W2ET  8192000UL
#define WS_W2DT  10289152UL
#define WS_W3DT  12386304UL
#define WS_H1    15597568UL
#define WS_H2    23986176UL
#define WS_D1    32374784UL
#define WS_D2    40763392UL
#define WS_Z     49152000UL
#define WS_ACC   49168384UL

extern "C" void kernel_launch(void* const* d_in, const int* in_sizes, int n_in,
                              void* d_out, int out_size, void* d_ws, size_t ws_size,
                              hipStream_t stream) {
  const float* x      = (const float*)d_in[0];
  const float* eps_z  = (const float*)d_in[1];
  const float* eps_pr = (const float*)d_in[2];
  const float* eW1 = (const float*)d_in[3];
  const float* eb1 = (const float*)d_in[4];
  const float* eW2 = (const float*)d_in[5];
  const float* eb2 = (const float*)d_in[6];
  const float* eW3 = (const float*)d_in[7];
  const float* eb3 = (const float*)d_in[8];
  const float* dW1 = (const float*)d_in[9];
  const float* db1 = (const float*)d_in[10];
  const float* dW2 = (const float*)d_in[11];
  const float* db2 = (const float*)d_in[12];
  const float* dW3 = (const float*)d_in[13];
  const float* db3 = (const float*)d_in[14];
  float* out = (float*)d_out;

  char* ws = (char*)d_ws;
  _Float16* xh   = (_Float16*)(ws + WS_XH);
  _Float16* w1t  = (_Float16*)(ws + WS_W1T);
  _Float16* w2et = (_Float16*)(ws + WS_W2ET);
  _Float16* w2dt = (_Float16*)(ws + WS_W2DT);
  _Float16* w3dt = (_Float16*)(ws + WS_W3DT);
  _Float16* h1   = (_Float16*)(ws + WS_H1);
  _Float16* h2   = (_Float16*)(ws + WS_H2);
  _Float16* d1   = (_Float16*)(ws + WS_D1);
  _Float16* d2   = (_Float16*)(ws + WS_D2);
  float*    zbuf = (float*)(ws + WS_Z);
  float*    acc  = (float*)(ws + WS_ACC);

  // --- prep ---
  k_zero3<<<1, 64, 0, stream>>>(acc);
  k_cvt_x<<<(NROWS * KP1) / 256, 256, 0, stream>>>(x, xh);
  k_transpose_w<<<(HD * KP1) / 256, 256, 0, stream>>>(eW1, w1t, IN_DIM, HD, KP1);
  k_transpose_w<<<(HD * HD) / 256, 256, 0, stream>>>(eW2, w2et, HD, HD, HD);
  k_transpose_w<<<(HD * HD) / 256, 256, 0, stream>>>(dW2, w2dt, HD, HD, HD);
  k_transpose_w<<<(OUT3 * HD) / 256, 256, 0, stream>>>(dW3, w3dt, HD, OUT3, HD);

  // --- encoder ---
  dim3 g1(HD / TN, NROWS / TM);
  k_gemm_tanh<<<g1, 256, 0, stream>>>(xh, w1t, eb1, h1, HD, KP1);
  k_gemm_tanh<<<g1, 256, 0, stream>>>(h1, w2et, eb2, h2, HD, HD);
  k_enc3<<<NROWS / 8, 256, 0, stream>>>(h2, eW3, eb3, eps_z, zbuf, out);

  // --- decoder ---
  k_dec1<<<(NROWS * HD) / 256, 256, 0, stream>>>(zbuf, dW1, db1, d1);
  k_gemm_tanh<<<g1, 256, 0, stream>>>(d1, w2dt, db2, d2, HD, HD);
  dim3 g3(IN_DIM / 16, NROWS / 128);   // 49 x 32
  k_dec3_rec<<<g3, 256, 0, stream>>>(d2, w3dt, db3, x, out);

  // --- MMD + finalize ---
  k_mmd<<<NROWS / 256, 256, 0, stream>>>(zbuf, eps_pr, acc);
  k_final<<<NROWS / 256, 256, 0, stream>>>(out, acc);

  (void)in_sizes; (void)n_in; (void)out_size; (void)ws_size;
}
